// CausalAttentionSingleHead_76175539962080
// MI455X (gfx1250) — matrix operand order repaired and checked by reading, not compiled, not run on hardware
//
#include <hip/hip_runtime.h>
#include <hip/hip_bf16.h>

// ---------------------------------------------------------------------------
// Causal single-head attention, S=4096, Dmodel=1024, Dk=128, on gfx1250.
// All matmuls via v_wmma_f32_16x16x32_bf16 (wave32 WMMA, f32 accumulate).
// Flash attention processes 64 keys/iteration to amortize softmax VALU work.
// ---------------------------------------------------------------------------

typedef __attribute__((ext_vector_type(16))) __bf16 v16bf;
typedef __attribute__((ext_vector_type(8)))  __bf16 v8bf;
typedef __attribute__((ext_vector_type(8)))  float  v8f;

#define SEQ     4096
#define DMODEL  1024
#define DK      128
#define MT      (SEQ / 16)      // 256 row tiles
#define KC      (DMODEL / 32)   // 32 k-chunks for projections
#define NT      (DK / 16)       // 8 col tiles of dk
#define BK      64              // keys per flash iteration

// A-fragment k-index map (16-bit A 16x32, ISA 7.12.2):
// lane half h, element e -> K = (e<8 ? h*8+e : 16 + h*8 + (e-8))
__device__ __forceinline__ int kmapA(int h, int e) {
    return (e < 8) ? (h * 8 + e) : (16 + h * 8 + (e - 8));
}

union FragU { v16bf v; v8bf half2[2]; };

// ---------------------------------------------------------------------------
// Pack X [S, DMODEL] f32 -> Xb bf16 in A-fragment order:
// dst[((mt*KC + kc)*32 + lane)*16 + e] = X[mt*16 + (lane&15), kc*32 + kmapA(lane>>4, e)]
// ---------------------------------------------------------------------------
__global__ void pack_x_kernel(const float* __restrict__ X, __bf16* __restrict__ Xb) {
    int tid = blockIdx.x * blockDim.x + threadIdx.x;
    if (tid >= SEQ * DMODEL) return;
    int e    = tid & 15;
    int lane = (tid >> 4) & 31;
    int f    = tid >> 9;          // fragment id = mt*KC + kc
    int kc   = f & (KC - 1);
    int mt   = f >> 5;
    int row  = mt * 16 + (lane & 15);
    int col  = kc * 32 + kmapA(lane >> 4, e);
    Xb[tid] = (__bf16)X[row * DMODEL + col];
}

// ---------------------------------------------------------------------------
// Pack W [DK, DMODEL] f32 -> Wb bf16 in B-fragment order (B = W^T, 32x16 tiles):
// dst[((kc*NT + nt)*32 + lane)*16 + e] = W[nt*16 + (lane&15), kc*32 + kmapA(lane>>4, e)]
// ---------------------------------------------------------------------------
__global__ void pack_w_kernel(const float* __restrict__ W, __bf16* __restrict__ Wb) {
    int tid = blockIdx.x * blockDim.x + threadIdx.x;
    if (tid >= DK * DMODEL) return;
    int e    = tid & 15;
    int lane = (tid >> 4) & 31;
    int g    = tid >> 9;          // fragment id = kc*NT + nt
    int nt   = g & (NT - 1);
    int kc   = g >> 3;
    int orow = nt * 16 + (lane & 15);          // output feature (dk index)
    int icol = kc * 32 + kmapA(lane >> 4, e);  // input feature (dmodel index)
    Wb[tid] = (__bf16)W[orow * DMODEL + icol];
}

// ---------------------------------------------------------------------------
// QKV projection: out = X @ W^T + b.
// One wave per 16x16 output tile column strip; block = 8 waves covers dk=128.
// mode 0/1: row-major bf16 out [S, DK] (Q, K).  mode 2: transposed Vt [DK, S].
// ---------------------------------------------------------------------------
__global__ void qkv_gemm_kernel(const __bf16* __restrict__ Xb,
                                const __bf16* __restrict__ Wb,
                                const float*  __restrict__ bias,
                                __bf16* __restrict__ out, int mode) {
    const int lane = threadIdx.x & 31;
    const int wave = threadIdx.x >> 5;   // 0..7 -> nt
    const int mt   = blockIdx.x;
    const int nt   = wave;
    const int n    = lane & 15;
    const int h    = lane >> 4;

    const float bcol = bias[nt * 16 + n];
    v8f c;
#pragma unroll
    for (int r = 0; r < 8; ++r) c[r] = bcol;

    const __bf16* ap = Xb + ((size_t)(mt * KC) * 32 + lane) * 16;
    const __bf16* bp = Wb + ((size_t)nt * 32 + lane) * 16;
#pragma unroll 4
    for (int kc = 0; kc < KC; ++kc) {
        v16bf a = *(const v16bf*)(ap + (size_t)kc * 512);            // (mt*KC+kc) frags
        v16bf b = *(const v16bf*)(bp + (size_t)kc * (NT * 512));     // (kc*NT+nt) frags
        c = __builtin_amdgcn_wmma_f32_16x16x32_bf16(false, a, false, b,
                                                    (short)0, c, false, false);
    }

    if (mode != 2) {
#pragma unroll
        for (int r = 0; r < 8; ++r) {
            int row = mt * 16 + r + 8 * h;
            out[(size_t)row * DK + nt * 16 + n] = (__bf16)c[r];
        }
    } else {
#pragma unroll
        for (int r = 0; r < 8; ++r) {
            int row = mt * 16 + r + 8 * h;
            out[(size_t)(nt * 16 + n) * SEQ + row] = (__bf16)c[r];
        }
    }
}

// ---------------------------------------------------------------------------
// Flash attention: 1 wave per 16-row query tile, 64 keys per iteration.
// Online softmax; P staged via LDS (C-layout f32 -> A-layout bf16).
// ---------------------------------------------------------------------------
__global__ void flash_attn_kernel(const __bf16* __restrict__ Qb,   // [S, DK] row-major
                                  const __bf16* __restrict__ Kb,   // [S, DK] row-major
                                  const __bf16* __restrict__ Vt,   // [DK, S] (transposed)
                                  float* __restrict__ O) {         // [S, DK]
    __shared__ __align__(16) __bf16 pbuf[4][16 * BK];   // per-wave P staging (8 KB)

    const int lane = threadIdx.x & 31;
    const int wave = threadIdx.x >> 5;                  // 4 waves / block
    const int qt   = blockIdx.x * 4 + wave;             // query tile 0..255
    const int n    = lane & 15;
    const int h    = lane >> 4;
    const float scale = 0.08838834764831845f;           // 1/sqrt(128)

    // Load Q A-fragments for this tile: 4 chunks of 16x32.
    FragU qa[4];
    {
        const __bf16* qrow = Qb + (size_t)(qt * 16 + n) * DK;
#pragma unroll
        for (int c = 0; c < 4; ++c) {
            qa[c].half2[0] = *(const v8bf*)(qrow + c * 32 + h * 16);
            qa[c].half2[1] = *(const v8bf*)(qrow + c * 32 + h * 16 + 8);
        }
    }

    v8f o[NT];
#pragma unroll
    for (int t = 0; t < NT; ++t) o[t] = (v8f){};
    float mrow[8], lrow[8];
#pragma unroll
    for (int r = 0; r < 8; ++r) { mrow[r] = -1e30f; lrow[r] = 0.0f; }

    const int niter = qt / 4 + 1;            // 64-key blocks up to the diagonal
    for (int it = 0; it < niter; ++it) {
        const int kbase = it * BK;

        // Prefetch next iteration's K rows and V panel into cache (TA runs ahead
        // of the WMMA pipeline; speculative, dropped if not needed).
        if (it + 1 < niter) {
            const __bf16* kn = Kb + (size_t)(kbase + BK + lane) * DK;
            __builtin_prefetch(kn, 0, 3);
            __builtin_prefetch(kn + (size_t)32 * DK, 0, 3);
#pragma unroll
            for (int q8 = 0; q8 < 4; ++q8)
                __builtin_prefetch(Vt + (size_t)(q8 * 32 + lane) * SEQ + kbase + BK, 0, 3);
        }

        // ---- scores: four 16x16 tiles over dk=128 (16 WMMAs) ----
        v8f s[4];
#pragma unroll
        for (int u = 0; u < 4; ++u) s[u] = (v8f){};
#pragma unroll
        for (int c = 0; c < 4; ++c) {
#pragma unroll
            for (int u = 0; u < 4; ++u) {
                v16bf kf = *(const v16bf*)(Kb + (size_t)(kbase + u * 16 + n) * DK
                                           + c * 32 + h * 16);
                s[u] = __builtin_amdgcn_wmma_f32_16x16x32_bf16(false, qa[c].v, false, kf,
                                                               (short)0, s[u], false, false);
            }
        }

        // ---- scale + causal mask + online softmax (per row, 64 cols) ----
#pragma unroll
        for (int r = 0; r < 8; ++r) {
            const int rowg = qt * 16 + r + 8 * h;
            float x[4];
#pragma unroll
            for (int u = 0; u < 4; ++u) {
                const int cg = kbase + u * 16 + n;
                x[u] = (cg <= rowg) ? s[u][r] * scale : -1e30f;
            }
            float mx = fmaxf(fmaxf(x[0], x[1]), fmaxf(x[2], x[3]));
#pragma unroll
            for (int d = 1; d < 16; d <<= 1) mx = fmaxf(mx, __shfl_xor(mx, d, 32));
            float mnew  = fmaxf(mrow[r], mx);
            float alpha = __expf(mrow[r] - mnew);
            float p[4];
#pragma unroll
            for (int u = 0; u < 4; ++u) p[u] = __expf(x[u] - mnew);
            float ps = (p[0] + p[1]) + (p[2] + p[3]);
#pragma unroll
            for (int d = 1; d < 16; d <<= 1) ps += __shfl_xor(ps, d, 32);
            lrow[r] = lrow[r] * alpha + ps;
            mrow[r] = mnew;
#pragma unroll
            for (int t = 0; t < NT; ++t) o[t][r] *= alpha;

            // stage P (C-layout) into LDS row-major 16x64
#pragma unroll
            for (int u = 0; u < 4; ++u)
                pbuf[wave][(r + 8 * h) * BK + u * 16 + n] = (__bf16)p[u];
        }

        // ---- re-read P as two 16x32 A-fragments (in-order LDS, same wave) ----
        FragU pa[2];
#pragma unroll
        for (int c2 = 0; c2 < 2; ++c2) {
            pa[c2].half2[0] = *(const v8bf*)&pbuf[wave][n * BK + c2 * 32 + h * 16];
            pa[c2].half2[1] = *(const v8bf*)&pbuf[wave][n * BK + c2 * 32 + h * 16 + 8];
        }

        // ---- O += P @ V over 64 keys (16 WMMAs) ----
#pragma unroll
        for (int c2 = 0; c2 < 2; ++c2) {
#pragma unroll
            for (int t = 0; t < NT; ++t) {
                v16bf vb = *(const v16bf*)(Vt + (size_t)(t * 16 + n) * SEQ
                                           + kbase + c2 * 32 + h * 16);
                o[t] = __builtin_amdgcn_wmma_f32_16x16x32_bf16(false, pa[c2].v, false, vb,
                                                               (short)0, o[t], false, false);
            }
        }
    }

    // ---- finalize: divide by row sums, store f32 ----
#pragma unroll
    for (int r = 0; r < 8; ++r) {
        const float rinv = 1.0f / lrow[r];
        const int   row  = qt * 16 + r + 8 * h;
#pragma unroll
        for (int t = 0; t < NT; ++t)
            O[(size_t)row * DK + t * 16 + n] = o[t][r] * rinv;
    }
}

// ---------------------------------------------------------------------------
// Host-side orchestration
// ---------------------------------------------------------------------------
extern "C" void kernel_launch(void* const* d_in, const int* in_sizes, int n_in,
                              void* d_out, int out_size, void* d_ws, size_t ws_size,
                              hipStream_t stream) {
    const float* X  = (const float*)d_in[0];
    const float* Wq = (const float*)d_in[1];
    const float* bq = (const float*)d_in[2];
    const float* Wk = (const float*)d_in[3];
    const float* bk = (const float*)d_in[4];
    const float* Wv = (const float*)d_in[5];
    const float* bv = (const float*)d_in[6];
    float* O = (float*)d_out;

    // Workspace layout (bytes), all power-of-two aligned:
    char* ws = (char*)d_ws;
    __bf16* Xb  = (__bf16*)(ws);                                  // 8 MB
    __bf16* Wbq = (__bf16*)(ws + (size_t)8 * 1024 * 1024);        // 256 KB
    __bf16* Wbk = (__bf16*)(ws + (size_t)8 * 1024 * 1024 + 262144);
    __bf16* Wbv = (__bf16*)(ws + (size_t)8 * 1024 * 1024 + 2 * 262144);
    __bf16* Qb  = (__bf16*)(ws + (size_t)8 * 1024 * 1024 + 3 * 262144);            // 1 MB
    __bf16* Kb  = (__bf16*)(ws + (size_t)8 * 1024 * 1024 + 3 * 262144 + 1048576);
    __bf16* Vt  = (__bf16*)(ws + (size_t)8 * 1024 * 1024 + 3 * 262144 + 2 * 1048576);

    // 1) pack activations & weights into WMMA fragment order (bf16)
    pack_x_kernel<<<(SEQ * DMODEL) / 256, 256, 0, stream>>>(X, Xb);
    pack_w_kernel<<<(DK * DMODEL) / 256, 256, 0, stream>>>(Wq, Wbq);
    pack_w_kernel<<<(DK * DMODEL) / 256, 256, 0, stream>>>(Wk, Wbk);
    pack_w_kernel<<<(DK * DMODEL) / 256, 256, 0, stream>>>(Wv, Wbv);

    // 2) QKV projections (Q,K row-major bf16; V transposed bf16)
    qkv_gemm_kernel<<<MT, 256, 0, stream>>>(Xb, Wbq, bq, Qb, 0);
    qkv_gemm_kernel<<<MT, 256, 0, stream>>>(Xb, Wbk, bk, Kb, 1);
    qkv_gemm_kernel<<<MT, 256, 0, stream>>>(Xb, Wbv, bv, Vt, 2);

    // 3) causal flash attention -> f32 output
    flash_attn_kernel<<<MT / 4, 128, 0, stream>>>(Qb, Kb, Vt, O);
}
